// EdgeClassifierGNN_58171037057327
// MI455X (gfx1250) — matrix operand order repaired and checked by reading, not compiled, not run on hardware
//
#include <hip/hip_runtime.h>
#include <hip/hip_bf16.h>

#define N_NODES 50000
#define N_EDGES 800000
#define D_NODE  128
#define D_EDGE  16
#define HID     64
#define EPSB    1e-5f

typedef _Float16 half_t;
typedef __attribute__((ext_vector_type(16))) _Float16 v16h;
typedef __attribute__((ext_vector_type(8)))  float    v8f;

// ---------- fragment helpers ----------
__device__ __forceinline__ v16h mk16(uint4 a, uint4 b) {
  union { uint4 u[2]; v16h h; } t;
  t.u[0] = a; t.u[1] = b;
  return t.h;
}
__device__ __forceinline__ uint4 ld16(const half_t* p) { return *(const uint4*)p; }

// B layout (16-bit, 32x16 KxN): lane n (0..15) = column n, K[k0..k0+7],K[k0+16..k0+23];
// lane n+16 = column n, K[k0+8..k0+15],K[k0+24..k0+31].
__device__ __forceinline__ v16h load_bfrag(const half_t* B, int kstride,
                                           int col, int k0, int hi) {
  const half_t* p = B + (long)col * kstride + k0 + (hi ? 8 : 0);
  return mk16(ld16(p), ld16(p + 16));
}

__device__ __forceinline__ v8f wmma_f16(v16h a, v16h b, v8f c) {
  return __builtin_amdgcn_wmma_f32_16x16x32_f16(false, a, false, b, (short)0, c,
                                                false, false);
}

// ---------- elementwise / prep kernels ----------
__global__ void k_cvt_h(const float* __restrict__ in, half_t* __restrict__ out, long n) {
  long i = (long)blockIdx.x * blockDim.x + threadIdx.x;
  if (i < n) out[i] = (half_t)in[i];
}

// Wm1 [128 x 144] f32 -> [128 x 160] f16 zero-padded
__global__ void k_cvt_wm1(const float* __restrict__ in, half_t* __restrict__ out) {
  int i = blockIdx.x * blockDim.x + threadIdx.x;
  if (i >= 128 * 160) return;
  int o = i / 160, k = i % 160;
  out[i] = (half_t)((k < 144) ? in[o * 144 + k] : 0.0f);
}

// Wm3 [2 x 64] f32 -> [16 x 64] f16 zero-padded rows
__global__ void k_cvt_wm3(const float* __restrict__ in, half_t* __restrict__ out) {
  int i = blockIdx.x * blockDim.x + threadIdx.x;
  if (i >= 16 * 64) return;
  out[i] = (half_t)((i < 128) ? in[i] : 0.0f);
}

__global__ void k_count(const long long* __restrict__ ei, float* __restrict__ cnt) {
  int e = blockIdx.x * blockDim.x + threadIdx.x;
  if (e < N_EDGES) atomicAdd(cnt + ei[N_EDGES + e], 1.0f);
}

template <int F>
__global__ void k_scatter(const float* __restrict__ x, const long long* __restrict__ ei,
                          float* __restrict__ agg) {
  constexpr int G = F / 4;
  long t = (long)blockIdx.x * blockDim.x + threadIdx.x;
  long e = t / G;
  int  p = (int)(t % G);
  if (e >= N_EDGES) return;
  long s = (long)ei[e], d = (long)ei[N_EDGES + e];
  float4 v = *(const float4*)(x + s * F + p * 4);
  float* o = agg + d * F + p * 4;
  atomicAdd(o + 0, v.x); atomicAdd(o + 1, v.y);
  atomicAdd(o + 2, v.z); atomicAdd(o + 3, v.w);
}

template <int F>
__global__ void k_mean_h(const float* __restrict__ agg, const float* __restrict__ cnt,
                         half_t* __restrict__ out) {
  long i = (long)blockIdx.x * blockDim.x + threadIdx.x;
  if (i >= (long)N_NODES * F) return;
  long r = i / F;
  out[i] = (half_t)(agg[i] / fmaxf(cnt[r], 1.0f));
}

// ---------- SAGE linear: out = A1h @ B1h^T + A2h @ B2h^T + bias ----------
// block = 256 threads = 8 waves; wave = one 16-row M tile; N = 64 (4 ntiles).
// Both weight matrices staged in LDS so WMMA B-fragments come from ds_load.
template <int K>
__global__ __launch_bounds__(256) void sage_gemm(
    const half_t* __restrict__ A1, const half_t* __restrict__ B1,
    const half_t* __restrict__ A2, const half_t* __restrict__ B2,
    const float* __restrict__ bias, float* __restrict__ out, int nrows) {
  constexpr int KS = K / 32;
  __shared__ __align__(16) half_t b1s[64 * K];
  __shared__ __align__(16) half_t b2s[64 * K];
  {
    constexpr int NU = 64 * K / 8;  // uint4 elements
    const uint4* s1 = (const uint4*)B1;
    const uint4* s2 = (const uint4*)B2;
    uint4* d1 = (uint4*)b1s;
    uint4* d2 = (uint4*)b2s;
    for (int i = threadIdx.x; i < NU; i += 256) { d1[i] = s1[i]; d2[i] = s2[i]; }
  }

  const int lane = threadIdx.x & 31;
  const int w    = threadIdx.x >> 5;
  const int hi   = lane >> 4;
  const int m    = lane & 15;
  const int rowbase = blockIdx.x * 128 + w * 16;
  int rm = rowbase + m;
  if (rm >= nrows) rm = nrows - 1;   // clamp (stores are guarded)

  // A layout: lane m holds K[c0..c0+7],K[c0+16..c0+23] with c0 = k0 + hi*8
  v16h a1[KS], a2[KS];
#pragma unroll
  for (int ks = 0; ks < KS; ++ks) {
    const int c0 = ks * 32 + hi * 8;
    const half_t* p1 = A1 + (long)rm * K + c0;
    const half_t* p2 = A2 + (long)rm * K + c0;
    a1[ks] = mk16(ld16(p1), ld16(p1 + 16));
    a2[ks] = mk16(ld16(p2), ld16(p2 + 16));
  }
  __syncthreads();

#pragma unroll
  for (int nt = 0; nt < 4; ++nt) {
    const int col = nt * 16 + m;
    v8f acc = {0.f, 0.f, 0.f, 0.f, 0.f, 0.f, 0.f, 0.f};
#pragma unroll
    for (int ks = 0; ks < KS; ++ks)
      acc = wmma_f16(a1[ks], load_bfrag(b1s, K, col, ks * 32, hi), acc);
#pragma unroll
    for (int ks = 0; ks < KS; ++ks)
      acc = wmma_f16(a2[ks], load_bfrag(b2s, K, col, ks * 32, hi), acc);
    const float bv = bias[col];
#pragma unroll
    for (int r = 0; r < 8; ++r) {
      const int rr = rowbase + r + hi * 8;   // C: vgpr r = row r (lanes<16) / r+8
      if (rr < nrows) out[(long)rr * 64 + col] = acc[r] + bv;
    }
  }
}

// ---------- BatchNorm ----------
__global__ __launch_bounds__(256) void k_bn_stats(const float* __restrict__ h,
                                                  float* __restrict__ sum,
                                                  float* __restrict__ sq) {
  __shared__ float ss[4][64], qq[4][64];
  const int c = threadIdx.x & 63, g = threadIdx.x >> 6;
  float s = 0.f, q = 0.f;
  const int r0 = blockIdx.x * 1024 + g;
  for (int j = 0; j < 256; ++j) {
    int r = r0 + j * 4;
    if (r < N_NODES) { float v = h[(long)r * 64 + c]; s += v; q += v * v; }
  }
  ss[g][c] = s; qq[g][c] = q;
  __syncthreads();
  if (g == 0) {
    atomicAdd(sum + c, ss[0][c] + ss[1][c] + ss[2][c] + ss[3][c]);
    atomicAdd(sq  + c, qq[0][c] + qq[1][c] + qq[2][c] + qq[3][c]);
  }
}

__global__ void k_bn_apply(const float* __restrict__ hraw,
                           const float* __restrict__ sum, const float* __restrict__ sq,
                           const float* __restrict__ gma, const float* __restrict__ bta,
                           float* __restrict__ hout, half_t* __restrict__ houth) {
  long i = (long)blockIdx.x * blockDim.x + threadIdx.x;
  if (i >= (long)N_NODES * 64) return;
  int c = (int)(i & 63);
  const float inv_n = 1.0f / (float)N_NODES;
  float mu  = sum[c] * inv_n;
  float var = sq[c] * inv_n - mu * mu;
  float v = (hraw[i] - mu) * rsqrtf(var + EPSB) * gma[c] + bta[c];
  v = fmaxf(v, 0.f);
  if (hout) hout[i] = v;
  houth[i] = (half_t)v;
}

// ---------- fused edge MLP: concat(h[src], ea, h[dst]) -> 128 -> 64 -> 2 ----------
// 128 edges/block (8 waves x 16 rows). K1 = 144 padded to 160. Weights staged
// in LDS; activation restaging is wave-local (no barriers after the weight one).
// All three layers on WMMA. z2 aliases z1 LDS (z1 consumed before z2 written).
__device__ __forceinline__ uint4 edge_chunk(const half_t* __restrict__ h2h,
                                            const half_t* __restrict__ eah,
                                            long s, long d, long e, int c0) {
  if (c0 < 64)  return ld16(h2h + s * 64 + c0);
  if (c0 < 80)  return ld16(eah + e * 16 + (c0 - 64));
  if (c0 < 144) return ld16(h2h + d * 64 + (c0 - 80));
  return make_uint4(0u, 0u, 0u, 0u);
}

__global__ __launch_bounds__(256) void edge_mlp(
    const half_t* __restrict__ h2h, const half_t* __restrict__ eah,
    const long long* __restrict__ ei,
    const half_t* __restrict__ Wm1, const float* __restrict__ bm1,   // [128 x 160]
    const half_t* __restrict__ Wm2, const float* __restrict__ bm2,   // [64 x 128]
    const half_t* __restrict__ Wm3, const float* __restrict__ bm3,   // [16 x 64]
    float* __restrict__ out) {
  __shared__ __align__(16) half_t w1s[128 * 160];   // 40 KB
  __shared__ __align__(16) half_t w2s[64 * 128];    // 16 KB
  __shared__ __align__(16) half_t w3s[16 * 64];     //  2 KB
  __shared__ __align__(16) half_t zbuf[8][16][128]; // 32 KB (z1; z2 reuses cols 0..63)

  // stage all MLP weights into LDS (one barrier for the whole kernel)
  {
    const uint4* s1 = (const uint4*)Wm1;
    uint4* d1 = (uint4*)w1s;
    for (int i = threadIdx.x; i < 2560; i += 256) d1[i] = s1[i];
    const uint4* s2 = (const uint4*)Wm2;
    uint4* d2 = (uint4*)w2s;
    for (int i = threadIdx.x; i < 1024; i += 256) d2[i] = s2[i];
    const uint4* s3 = (const uint4*)Wm3;
    uint4* d3 = (uint4*)w3s;
    if (threadIdx.x < 128) d3[threadIdx.x] = s3[threadIdx.x];
  }

  const int lane = threadIdx.x & 31;
  const int w    = threadIdx.x >> 5;
  const int hi   = lane >> 4;
  const int m    = lane & 15;
  const long erow = (long)blockIdx.x * 128 + w * 16 + m;
  const long s = (long)ei[erow];
  const long d = (long)ei[(long)N_EDGES + erow];

  // Layer 1 A fragments (global gathers), K = 160 (5 ksteps)
  v16h a[5];
#pragma unroll
  for (int ks = 0; ks < 5; ++ks) {
    const int c0 = ks * 32 + hi * 8;
    a[ks] = mk16(edge_chunk(h2h, eah, s, d, erow, c0),
                 edge_chunk(h2h, eah, s, d, erow, c0 + 16));
  }
  __syncthreads();

  // Layer 1: N = 128 (8 ntiles), B from LDS
#pragma unroll
  for (int nt = 0; nt < 8; ++nt) {
    const int col = nt * 16 + m;
    v8f acc = {0.f, 0.f, 0.f, 0.f, 0.f, 0.f, 0.f, 0.f};
#pragma unroll
    for (int ks = 0; ks < 5; ++ks)
      acc = wmma_f16(a[ks], load_bfrag(w1s, 160, col, ks * 32, hi), acc);
    const float bv = bm1[col];
#pragma unroll
    for (int r = 0; r < 8; ++r)
      zbuf[w][r + hi * 8][col] = (half_t)fmaxf(acc[r] + bv, 0.f);
  }

  // Layer 2: K = 128 (4 ksteps), N = 64 (4 ntiles); A restaged from z1 (wave-local)
  v16h a2[4];
#pragma unroll
  for (int ks = 0; ks < 4; ++ks) {
    const int c0 = ks * 32 + hi * 8;
    const half_t* p = &zbuf[w][m][c0];
    a2[ks] = mk16(ld16(p), ld16(p + 16));
  }
#pragma unroll
  for (int nt = 0; nt < 4; ++nt) {
    const int col = nt * 16 + m;
    v8f acc = {0.f, 0.f, 0.f, 0.f, 0.f, 0.f, 0.f, 0.f};
#pragma unroll
    for (int ks = 0; ks < 4; ++ks)
      acc = wmma_f16(a2[ks], load_bfrag(w2s, 128, col, ks * 32, hi), acc);
    const float bv = bm2[col];
#pragma unroll
    for (int r = 0; r < 8; ++r)   // z2: cols 0..63, aliases z1 (already consumed)
      zbuf[w][r + hi * 8][col] = (half_t)fmaxf(acc[r] + bv, 0.f);
  }

  // Layer 3 on WMMA: K = 64 (2 ksteps), N = 16 (cols 0..1 valid)
  v16h a3[2];
#pragma unroll
  for (int ks = 0; ks < 2; ++ks) {
    const int c0 = ks * 32 + hi * 8;
    const half_t* p = &zbuf[w][m][c0];
    a3[ks] = mk16(ld16(p), ld16(p + 16));
  }
  v8f acc3 = {0.f, 0.f, 0.f, 0.f, 0.f, 0.f, 0.f, 0.f};
  acc3 = wmma_f16(a3[0], load_bfrag(w3s, 64, m, 0, hi), acc3);
  acc3 = wmma_f16(a3[1], load_bfrag(w3s, 64, m, 32, hi), acc3);
  if (m < 2) {
    const float bv = bm3[m];
#pragma unroll
    for (int r = 0; r < 8; ++r) {
      const long e = (long)blockIdx.x * 128 + w * 16 + r + hi * 8;
      out[e * 2 + m] = acc3[r] + bv;
    }
  }
}

// ---------- launch ----------
extern "C" void kernel_launch(void* const* d_in, const int* in_sizes, int n_in,
                              void* d_out, int out_size, void* d_ws, size_t ws_size,
                              hipStream_t stream) {
  const float*     x   = (const float*)d_in[0];
  const long long* ei  = (const long long*)d_in[1];
  const float*     ea  = (const float*)d_in[2];
  const float* W1l = (const float*)d_in[3];
  const float* b1l = (const float*)d_in[4];
  const float* W1r = (const float*)d_in[5];
  const float* g1  = (const float*)d_in[6];
  const float* be1 = (const float*)d_in[7];
  const float* W2l = (const float*)d_in[8];
  const float* b2l = (const float*)d_in[9];
  const float* W2r = (const float*)d_in[10];
  const float* g2  = (const float*)d_in[11];
  const float* be2 = (const float*)d_in[12];
  const float* Wm1 = (const float*)d_in[13];
  const float* bm1 = (const float*)d_in[14];
  const float* Wm2 = (const float*)d_in[15];
  const float* bm2 = (const float*)d_in[16];
  const float* Wm3 = (const float*)d_in[17];
  const float* bm3 = (const float*)d_in[18];
  float* out = (float*)d_out;

  char* ws = (char*)d_ws;
  size_t off = 0;
  auto alloc = [&](size_t bytes) {
    void* p = ws + off;
    off = (off + bytes + 255) & ~(size_t)255;
    return p;
  };

  float*  agg1   = (float*)alloc((size_t)N_NODES * 128 * 4);
  float*  agg2   = (float*)alloc((size_t)N_NODES * 64 * 4);
  float*  cnt    = (float*)alloc((size_t)N_NODES * 4);
  float*  h1     = (float*)alloc((size_t)N_NODES * 64 * 4);   // raw, then in-place BN
  float*  h2raw  = (float*)alloc((size_t)N_NODES * 64 * 4);
  half_t* xh     = (half_t*)alloc((size_t)N_NODES * 128 * 2);
  half_t* aggm1h = (half_t*)alloc((size_t)N_NODES * 128 * 2);
  half_t* h1h    = (half_t*)alloc((size_t)N_NODES * 64 * 2);
  half_t* aggm2h = (half_t*)alloc((size_t)N_NODES * 64 * 2);
  half_t* h2h    = (half_t*)alloc((size_t)N_NODES * 64 * 2);
  half_t* eah    = (half_t*)alloc((size_t)N_EDGES * 16 * 2);
  half_t* W1lh   = (half_t*)alloc(64 * 128 * 2);
  half_t* W1rh   = (half_t*)alloc(64 * 128 * 2);
  half_t* W2lh   = (half_t*)alloc(64 * 64 * 2);
  half_t* W2rh   = (half_t*)alloc(64 * 64 * 2);
  half_t* Wm1h   = (half_t*)alloc(128 * 160 * 2);
  half_t* Wm2h   = (half_t*)alloc(64 * 128 * 2);
  half_t* Wm3h   = (half_t*)alloc(16 * 64 * 2);
  float*  stats  = (float*)alloc(4 * 64 * 4);  // sum1, sq1, sum2, sq2
  float *sum1 = stats, *sq1 = stats + 64, *sum2 = stats + 128, *sq2 = stats + 192;
  (void)in_sizes; (void)n_in; (void)out_size; (void)ws_size;

  // zero accumulation buffers (capturable)
  hipMemsetAsync(agg1, 0, (size_t)N_NODES * 128 * 4, stream);
  hipMemsetAsync(agg2, 0, (size_t)N_NODES * 64 * 4, stream);
  hipMemsetAsync(cnt, 0, (size_t)N_NODES * 4, stream);
  hipMemsetAsync(stats, 0, 4 * 64 * 4, stream);

  const int T = 256;
  auto blks = [](long n, int t) { return (int)((n + t - 1) / t); };

  // f16 conversions
  k_cvt_h<<<blks((long)N_NODES * 128, T), T, 0, stream>>>(x, xh, (long)N_NODES * 128);
  k_cvt_h<<<blks(64 * 128, T), T, 0, stream>>>(W1l, W1lh, 64 * 128);
  k_cvt_h<<<blks(64 * 128, T), T, 0, stream>>>(W1r, W1rh, 64 * 128);
  k_cvt_h<<<blks(64 * 64, T), T, 0, stream>>>(W2l, W2lh, 64 * 64);
  k_cvt_h<<<blks(64 * 64, T), T, 0, stream>>>(W2r, W2rh, 64 * 64);
  k_cvt_h<<<blks(64 * 128, T), T, 0, stream>>>(Wm2, Wm2h, 64 * 128);
  k_cvt_wm1<<<blks(128 * 160, T), T, 0, stream>>>(Wm1, Wm1h);
  k_cvt_wm3<<<blks(16 * 64, T), T, 0, stream>>>(Wm3, Wm3h);
  k_cvt_h<<<blks((long)N_EDGES * 16, T), T, 0, stream>>>(ea, eah, (long)N_EDGES * 16);

  // degrees + layer-1 aggregation
  k_count<<<blks(N_EDGES, T), T, 0, stream>>>(ei, cnt);
  k_scatter<128><<<blks((long)N_EDGES * 32, T), T, 0, stream>>>(x, ei, agg1);
  k_mean_h<128><<<blks((long)N_NODES * 128, T), T, 0, stream>>>(agg1, cnt, aggm1h);

  // SAGE1 + BN1 + ReLU
  const int gemm_blocks = (N_NODES + 127) / 128;
  sage_gemm<128><<<gemm_blocks, 256, 0, stream>>>(aggm1h, W1lh, xh, W1rh, b1l, h1, N_NODES);
  k_bn_stats<<<(N_NODES + 1023) / 1024, 256, 0, stream>>>(h1, sum1, sq1);
  k_bn_apply<<<blks((long)N_NODES * 64, T), T, 0, stream>>>(h1, sum1, sq1, g1, be1, h1, h1h);

  // layer-2 aggregation + SAGE2 + BN2 + ReLU (f16 out only)
  k_scatter<64><<<blks((long)N_EDGES * 16, T), T, 0, stream>>>(h1, ei, agg2);
  k_mean_h<64><<<blks((long)N_NODES * 64, T), T, 0, stream>>>(agg2, cnt, aggm2h);
  sage_gemm<64><<<gemm_blocks, 256, 0, stream>>>(aggm2h, W2lh, h1h, W2rh, b2l, h2raw, N_NODES);
  k_bn_stats<<<(N_NODES + 1023) / 1024, 256, 0, stream>>>(h2raw, sum2, sq2);
  k_bn_apply<<<blks((long)N_NODES * 64, T), T, 0, stream>>>(h2raw, sum2, sq2, g2, be2,
                                                            (float*)nullptr, h2h);

  // fused edge MLP (all-WMMA) -> logits
  edge_mlp<<<N_EDGES / 128, 256, 0, stream>>>(h2h, eah, ei, Wm1h, bm1, Wm2h, bm2,
                                              Wm3h, bm3, out);
}
